// EncoderGraphUNet_70136815943928
// MI455X (gfx1250) — matrix operand order repaired and checked by reading, not compiled, not run on hardware
//
#include <hip/hip_runtime.h>
#include <math.h>

typedef __bf16 bf16;
typedef __attribute__((ext_vector_type(16))) __bf16 v16bf;
typedef __attribute__((ext_vector_type(8)))  __bf16 v8bf;
typedef __attribute__((ext_vector_type(8)))  float  v8f;

#define HID 16
#define N0  4096

// ---------------- build adjacency ----------------
__global__ void k_scatter_edges(const int* __restrict__ ei, float* __restrict__ A,
                                int e2, int n) {
  int e = blockIdx.x * blockDim.x + threadIdx.x;
  if (e >= e2) return;
  int s = ei[e];
  int d = ei[e2 + e];
  atomicAdd(&A[(size_t)d * n + s], 1.0f);  // integer counts: order-independent
}

__global__ void k_f2bf(const float* __restrict__ s, bf16* __restrict__ d, size_t cnt) {
  size_t i = (size_t)blockIdx.x * blockDim.x + threadIdx.x;
  if (i < cnt) d[i] = (bf16)s[i];
}

// B = A + I  (diag(A)==0 at every level in this pipeline)
__global__ void k_make_B(const bf16* __restrict__ A, bf16* __restrict__ B, int n) {
  size_t i = (size_t)blockIdx.x * blockDim.x + threadIdx.x;
  size_t cnt = (size_t)n * n;
  if (i >= cnt) return;
  int r = (int)(i / (size_t)n);
  int c = (int)(i - (size_t)r * n);
  bf16 v = A[i];
  if (r == c) v = (bf16)1.0f;
  B[i] = v;
}

// ---------------- C = B @ B (B symmetric, bf16) via WMMA ----------------
__device__ __forceinline__ v16bf load_afrag(const bf16* rowp, int kb, int aHalf) {
  const bf16* p = rowp + kb + aHalf;       // A frag: K interleave 0-7/16-23 | 8-15/24-31
  v8bf lo = *(const v8bf*)(p);
  v8bf hi = *(const v8bf*)(p + 16);
  return __builtin_shufflevector(lo, hi, 0, 1, 2, 3, 4, 5, 6, 7,
                                 8, 9, 10, 11, 12, 13, 14, 15);
}

__device__ __forceinline__ v16bf load_bfrag(const bf16* rowp, int kb, int bHalf) {
  const bf16* p = rowp + kb + bHalf;       // B frag: K interleave 0-7/16-23 | 8-15/24-31
  v8bf lo = *(const v8bf*)(p);
  v8bf hi = *(const v8bf*)(p + 16);
  return __builtin_shufflevector(lo, hi, 0, 1, 2, 3, 4, 5, 6, 7,
                                 8, 9, 10, 11, 12, 13, 14, 15);
}

// block: 256 thr = 8 waves, covers 128x128 of C. wave -> 64x32 (4x2 tiles of 16x16).
// K-loop unrolled x2 with double-buffered fragments to overlap loads with WMMA.
__global__ __launch_bounds__(256) void k_spgemm_wmma(const bf16* __restrict__ B,
                                                     float* __restrict__ C, int n) {
  const int lane  = threadIdx.x & 31;
  const int wave  = threadIdx.x >> 5;
  const int wr    = (wave >> 2) * 64;   // 2 wave-rows
  const int wc    = (wave & 3) * 32;    // 4 wave-cols
  const int br    = blockIdx.y * 128;
  const int bc    = blockIdx.x * 128;
  const int lr    = lane & 15;
  const int aHalf = (lane >> 4) * 8;
  const int bHalf = (lane >> 4) * 8;

  v8f acc[4][2];
#pragma unroll
  for (int tr = 0; tr < 4; ++tr)
#pragma unroll
    for (int tc = 0; tc < 2; ++tc) acc[tr][tc] = (v8f){0, 0, 0, 0, 0, 0, 0, 0};

  const bf16* ap[4];
  const bf16* bp[2];
#pragma unroll
  for (int tr = 0; tr < 4; ++tr)
    ap[tr] = B + (size_t)(br + wr + tr * 16 + lr) * n;
#pragma unroll
  for (int tc = 0; tc < 2; ++tc)   // B[k, col] == B[col, k] (symmetric) -> row read
    bp[tc] = B + (size_t)(bc + wc + tc * 16 + lr) * n;

  v16bf af0[4], bf0[2], af1[4], bf1[2];
#pragma unroll
  for (int tr = 0; tr < 4; ++tr) af0[tr] = load_afrag(ap[tr], 0, aHalf);
#pragma unroll
  for (int tc = 0; tc < 2; ++tc) bf0[tc] = load_bfrag(bp[tc], 0, bHalf);

  for (int kb = 0; kb < n; kb += 64) {  // n % 64 == 0 for all levels
    // prefetch k-step kb+32 into buffer 1 (always in range)
#pragma unroll
    for (int tr = 0; tr < 4; ++tr) af1[tr] = load_afrag(ap[tr], kb + 32, aHalf);
#pragma unroll
    for (int tc = 0; tc < 2; ++tc) bf1[tc] = load_bfrag(bp[tc], kb + 32, bHalf);
    // consume buffer 0
#pragma unroll
    for (int tr = 0; tr < 4; ++tr)
#pragma unroll
      for (int tc = 0; tc < 2; ++tc)
        acc[tr][tc] = __builtin_amdgcn_wmma_f32_16x16x32_bf16(
            false, af0[tr], false, bf0[tc], (short)0, acc[tr][tc], false, false);
    // prefetch k-step kb+64 into buffer 0
    if (kb + 64 < n) {
#pragma unroll
      for (int tr = 0; tr < 4; ++tr) af0[tr] = load_afrag(ap[tr], kb + 64, aHalf);
#pragma unroll
      for (int tc = 0; tc < 2; ++tc) bf0[tc] = load_bfrag(bp[tc], kb + 64, bHalf);
    }
    // consume buffer 1
#pragma unroll
    for (int tr = 0; tr < 4; ++tr)
#pragma unroll
      for (int tc = 0; tc < 2; ++tc)
        acc[tr][tc] = __builtin_amdgcn_wmma_f32_16x16x32_bf16(
            false, af1[tr], false, bf1[tc], (short)0, acc[tr][tc], false, false);
  }

  const int rsub = (lane >> 4) * 8;  // C/D layout: M = v + 8*lanehi, N = lane&15
#pragma unroll
  for (int tr = 0; tr < 4; ++tr) {
    int gr0 = br + wr + tr * 16 + rsub;
#pragma unroll
    for (int tc = 0; tc < 2; ++tc) {
      int gc = bc + wc + tc * 16 + (lane & 15);
#pragma unroll
      for (int v = 0; v < 8; ++v)
        C[(size_t)(gr0 + v) * n + gc] = acc[tr][tc][v];
    }
  }
}

// ---------------- GCN pieces ----------------
// dinv[i] = 1/sqrt(rowsum(A) + 2)   (improved=True adds 2I; diag(A)==0)
__global__ void k_row_deg(const bf16* __restrict__ A, int n, float* __restrict__ dinv) {
  int wave = threadIdx.x >> 5, lane = threadIdx.x & 31;
  int row = blockIdx.x * 8 + wave;
  float s = 0.f;
  const bf16* ar = A + (size_t)row * n;
  for (int j = lane; j < n; j += 32) s += (float)ar[j];
  for (int o = 16; o > 0; o >>= 1) s += __shfl_xor(s, o, 32);
  if (lane == 0) dinv[row] = rsqrtf(s + 2.0f);
}

__global__ void k_dense_xw(const float* __restrict__ x, const float* __restrict__ W,
                           float* __restrict__ z, int n, int ic) {
  int t = blockIdx.x * blockDim.x + threadIdx.x;
  if (t >= n * HID) return;
  int i = t >> 4, c = t & 15;
  float s = 0.f;
  for (int k = 0; k < ic; ++k) s += x[(size_t)i * ic + k] * W[k * HID + c];
  z[t] = s;
}

// out[i,c] = dinv[i]*( sum_j A[i,j]*dinv[j]*z[j,c] + 2*dinv[i]*z[i,c] ) + b ; opt relu
__global__ __launch_bounds__(256) void k_spmm(const bf16* __restrict__ A, int n,
                                              const float* __restrict__ z,
                                              const float* __restrict__ dinv,
                                              const float* __restrict__ bias,
                                              float* __restrict__ out, int relu) {
  __shared__ float zs[128 * HID];
  int tx = threadIdx.x & 15;   // col
  int ty = threadIdx.x >> 4;   // row in 16-row tile
  int row = blockIdx.x * 16 + ty;
  float acc = 0.f;
  for (int jc = 0; jc < n; jc += 128) {
    for (int t = threadIdx.x; t < 128 * HID; t += 256) {
      int rr = t >> 4;
      zs[t] = z[(size_t)(jc + rr) * HID + (t & 15)] * dinv[jc + rr];
    }
    __syncthreads();
    const bf16* ar = A + (size_t)row * n + jc;
#pragma unroll 8
    for (int jj = 0; jj < 128; ++jj) acc += (float)ar[jj] * zs[jj * HID + tx];
    __syncthreads();
  }
  float self = z[(size_t)row * HID + tx] * dinv[row];
  float v = dinv[row] * (acc + 2.0f * self) + bias[tx];
  if (relu && v < 0.f) v = 0.f;
  out[(size_t)row * HID + tx] = v;
}

// ---------------- top-k pooling ----------------
__global__ void k_score(const float* __restrict__ x, const float* __restrict__ p,
                        float* __restrict__ sc, int n) {
  int i = blockIdx.x * blockDim.x + threadIdx.x;
  if (i >= n) return;
  float pn = 0.f;
  for (int c = 0; c < HID; ++c) pn += p[c] * p[c];
  pn = sqrtf(pn);
  float s = 0.f;
  for (int c = 0; c < HID; ++c) s += x[(size_t)i * HID + c] * p[c];
  sc[i] = tanhf(s / pn);
}

// single-block bitonic sort, descending by score, ties -> lower index first
__global__ __launch_bounds__(1024) void k_topk_sort(const float* __restrict__ score,
                                                    int n, int k, int* __restrict__ perm,
                                                    float* __restrict__ psc) {
  __shared__ float skey[4096];
  __shared__ int sidx[4096];
  for (int i = threadIdx.x; i < n; i += blockDim.x) { skey[i] = score[i]; sidx[i] = i; }
  __syncthreads();
  for (int ksz = 2; ksz <= n; ksz <<= 1) {
    for (int jsz = ksz >> 1; jsz >= 1; jsz >>= 1) {
      for (int e = threadIdx.x; e < n; e += blockDim.x) {
        int partner = e ^ jsz;
        if (partner > e) {
          bool desc = ((e & ksz) == 0);
          float sa = skey[e], sb = skey[partner];
          int ia = sidx[e], ib = sidx[partner];
          bool aFirst = (sa > sb) || (sa == sb && ia < ib);
          bool doSwap = desc ? !aFirst : aFirst;
          if (doSwap) { skey[e] = sb; skey[partner] = sa; sidx[e] = ib; sidx[partner] = ia; }
        }
      }
      __syncthreads();
    }
  }
  for (int i = threadIdx.x; i < k; i += blockDim.x) { perm[i] = sidx[i]; psc[i] = skey[i]; }
}

__global__ void k_pool_x(const float* __restrict__ xin, const int* __restrict__ perm,
                         const float* __restrict__ psc, float* __restrict__ xo, int k) {
  int t = blockIdx.x * blockDim.x + threadIdx.x;
  if (t >= k * HID) return;
  int i = t >> 4, c = t & 15;
  xo[t] = xin[(size_t)perm[i] * HID + c] * psc[i];
}

__global__ void k_pool_A(const float* __restrict__ A2, const int* __restrict__ perm,
                         bf16* __restrict__ Ao, int k, int n) {
  size_t t = (size_t)blockIdx.x * blockDim.x + threadIdx.x;
  if (t >= (size_t)k * k) return;
  int i = (int)(t / (size_t)k), j = (int)(t - (size_t)i * k);
  float v = (i == j) ? 0.f : A2[(size_t)perm[i] * n + perm[j]];
  Ao[t] = (bf16)v;
}

// ---------------- up-path helpers ----------------
__global__ void k_copy(const float* __restrict__ s, float* __restrict__ d, int cnt) {
  int t = blockIdx.x * blockDim.x + threadIdx.x;
  if (t < cnt) d[t] = s[t];
}

__global__ void k_scatter_add(float* __restrict__ buf, const float* __restrict__ xc,
                              const int* __restrict__ perm, int k) {
  int t = blockIdx.x * blockDim.x + threadIdx.x;
  if (t >= k * HID) return;
  int i = t >> 4, c = t & 15;
  buf[(size_t)perm[i] * HID + c] += xc[t];  // perm rows distinct -> race-free
}

__global__ void k_normalize(const float* __restrict__ x, float* __restrict__ o, int n) {
  int i = blockIdx.x * blockDim.x + threadIdx.x;
  if (i >= n) return;
  float s = 0.f;
  for (int c = 0; c < HID; ++c) { float v = x[(size_t)i * HID + c]; s += v * v; }
  float d = fmaxf(sqrtf(s), 1e-3f);
  for (int c = 0; c < HID; ++c) o[(size_t)i * HID + c] = x[(size_t)i * HID + c] / d;
}

// ---------------- host orchestration ----------------
static void run_gcn(hipStream_t st, const bf16* A, int n, const float* xin, int ic,
                    const float* W, const float* b, int relu, float* out,
                    float* Z, float* DINV) {
  k_row_deg<<<n / 8, 256, 0, st>>>(A, n, DINV);
  k_dense_xw<<<(n * HID + 255) / 256, 256, 0, st>>>(xin, W, Z, n, ic);
  k_spmm<<<n / 16, 256, 0, st>>>(A, n, Z, DINV, b, out, relu);
}

extern "C" void kernel_launch(void* const* d_in, const int* in_sizes, int n_in,
                              void* d_out, int out_size, void* d_ws, size_t ws_size,
                              hipStream_t stream) {
  const float* x0 = (const float*)d_in[0];
  const int* ei = (const int*)d_in[1];
  const float* dW[5] = {(const float*)d_in[2], (const float*)d_in[3], (const float*)d_in[4],
                        (const float*)d_in[5], (const float*)d_in[6]};
  const float* db[5] = {(const float*)d_in[7], (const float*)d_in[8], (const float*)d_in[9],
                        (const float*)d_in[10], (const float*)d_in[11]};
  const float* pw[4] = {(const float*)d_in[12], (const float*)d_in[13],
                        (const float*)d_in[14], (const float*)d_in[15]};
  const float* uW[4] = {(const float*)d_in[16], (const float*)d_in[17],
                        (const float*)d_in[18], (const float*)d_in[19]};
  const float* ub[4] = {(const float*)d_in[20], (const float*)d_in[21],
                        (const float*)d_in[22], (const float*)d_in[23]};
  (void)n_in; (void)out_size; (void)ws_size;

  const int e2 = in_sizes[1] / 2;  // 131072 directed edges
  const int sizes[5] = {4096, 2048, 1024, 512, 256};

  // workspace carve (256B aligned)
  size_t off = 0;
  auto alloc = [&](size_t bytes) -> char* {
    off = (off + 255) & ~(size_t)255;
    char* p = (char*)d_ws + off;
    off += bytes;
    return p;
  };
  float* SCRF = (float*)alloc((size_t)N0 * N0 * 4);  // scatter target / B@B output
  bf16*  SCRB = (bf16*)alloc((size_t)N0 * N0 * 2);   // B = A + I
  bf16*  Alv[5];
  for (int l = 0; l < 5; ++l) Alv[l] = (bf16*)alloc((size_t)sizes[l] * sizes[l] * 2);
  float* XS[4];
  for (int l = 0; l < 4; ++l) XS[l] = (float*)alloc((size_t)sizes[l] * HID * 4);
  float* XCUR = (float*)alloc((size_t)N0 * HID * 4);
  float* XNEXT = (float*)alloc((size_t)N0 * HID * 4);
  float* XIN = (float*)alloc((size_t)N0 * HID * 4);
  float* Z = (float*)alloc((size_t)N0 * HID * 4);
  float* DINV = (float*)alloc((size_t)N0 * 4);
  float* SCORE = (float*)alloc((size_t)N0 * 4);
  float* PSC = (float*)alloc((size_t)2048 * 4);
  int* PERM[4];
  for (int l = 0; l < 4; ++l) PERM[l] = (int*)alloc((size_t)sizes[l + 1] * 4);

  // ---- build A0 ----
  hipMemsetAsync(SCRF, 0, (size_t)N0 * N0 * 4, stream);
  k_scatter_edges<<<(e2 + 255) / 256, 256, 0, stream>>>(ei, SCRF, e2, N0);
  k_f2bf<<<(unsigned)(((size_t)N0 * N0 + 255) / 256), 256, 0, stream>>>(
      SCRF, Alv[0], (size_t)N0 * N0);

  // ---- down path ----
  run_gcn(stream, Alv[0], N0, x0, 128, dW[0], db[0], 1, XS[0], Z, DINV);
  for (int i = 1; i <= 4; ++i) {
    int nn = sizes[i - 1], kk = sizes[i];
    size_t nn2 = (size_t)nn * nn;
    k_make_B<<<(unsigned)((nn2 + 255) / 256), 256, 0, stream>>>(Alv[i - 1], SCRB, nn);
    k_spgemm_wmma<<<dim3(nn / 128, nn / 128), 256, 0, stream>>>(SCRB, SCRF, nn);
    k_score<<<(nn + 255) / 256, 256, 0, stream>>>(XS[i - 1], pw[i - 1], SCORE, nn);
    k_topk_sort<<<1, 1024, 0, stream>>>(SCORE, nn, kk, PERM[i - 1], PSC);
    k_pool_x<<<(kk * HID + 255) / 256, 256, 0, stream>>>(XS[i - 1], PERM[i - 1], PSC, XIN, kk);
    k_pool_A<<<(unsigned)(((size_t)kk * kk + 255) / 256), 256, 0, stream>>>(
        SCRF, PERM[i - 1], Alv[i], kk, nn);
    float* outx = (i < 4) ? XS[i] : XCUR;
    run_gcn(stream, Alv[i], kk, XIN, HID, dW[i], db[i], 1, outx, Z, DINV);
  }

  // ---- up path ----
  float* cur = XCUR;
  for (int i = 0; i < 4; ++i) {
    int j = 3 - i;
    int nj = sizes[j], kj = sizes[j + 1];
    k_copy<<<(nj * HID + 255) / 256, 256, 0, stream>>>(XS[j], XIN, nj * HID);
    k_scatter_add<<<(kj * HID + 255) / 256, 256, 0, stream>>>(XIN, cur, PERM[j], kj);
    float* outx = (i & 1) ? XCUR : XNEXT;
    run_gcn(stream, Alv[j], nj, XIN, HID, uW[i], ub[i], (i < 3) ? 1 : 0, outx, Z, DINV);
    cur = outx;
  }

  // ---- F.normalize(out, p=2, dim=1, eps=1e-3) ----
  k_normalize<<<(N0 + 255) / 256, 256, 0, stream>>>(cur, (float*)d_out, N0);
}
